// SelfAttention_24833500905488
// MI455X (gfx1250) — compile-verified, matrix operand order checked by reading, not run on hardware
//
#include <hip/hip_runtime.h>
#include <math.h>

// Problem constants (match reference): B=4, T=2048, D=1024, single head.
#define BB 4
#define TT 2048
#define DD 1024

typedef __attribute__((ext_vector_type(16))) _Float16 v16h;
typedef __attribute__((ext_vector_type(8)))  float    v8f;

// TDM descriptor vector types (6-arg clang-23 builtin:
//   (u32x4 g0, i32x8 g1, i32x4, i32x4, i32x8, i32 cpol)).
typedef __attribute__((ext_vector_type(4))) unsigned tdm_u4;
typedef __attribute__((ext_vector_type(8))) int      tdm_i8;
typedef __attribute__((ext_vector_type(4))) int      tdm_i4;

// ---------------------------------------------------------------------------
// v_wmma_f32_16x16x32_f16 wrapper (codegen-confirmed builtin on gfx1250).
// ---------------------------------------------------------------------------
__device__ __forceinline__ v8f wmma16(v16h a, v16h b, v8f c) {
  return __builtin_amdgcn_wmma_f32_16x16x32_f16(
      /*neg_a=*/false, a, /*neg_b=*/false, b,
      /*c_mod=*/(short)0, c, /*reuse_a=*/false, /*reuse_b=*/false);
}

// ---------------------------------------------------------------------------
// Fragment fetch from an LDS f16 tile: two 16B chunks -> 2x ds_load_b128.
// CDNA5 16-bit A/B layout (wave32): lane%16 = row, kb=(lane>>4)*8:
//   elems 0..7  = K kb..kb+7,   elems 8..15 = K kb+16..kb+23.
// ---------------------------------------------------------------------------
__device__ __forceinline__ v16h ldfrag(const _Float16* p, int kb) {
  union { v16h v; float4 f[2]; } u;
  u.f[0] = *(const float4*)(p + kb);
  u.f[1] = *(const float4*)(p + kb + 16);
  return u.v;
}

// ---------------------------------------------------------------------------
// Tensor Data Mover: one 2D tile (128 rows x 32 halfwords) global -> LDS.
// D# group1 (shared by all 4 tiles; K = row pitch in elements):
//   data_size=1 (2B), pad_enable, pad_interval=3 (after 16 DWORDs = 64B row),
//   pad_amount=3 (4 DWORDs = 16B)  => LDS pitch 80B = 40 halfwords.
//   tensor_dim0=K, tensor_dim1=2^20 (ample), tile_dim0=32, tile_dim1=128.
// ---------------------------------------------------------------------------
__device__ __forceinline__ tdm_i8 tdm_make_g1(int K) {
  tdm_i8 g;
  g[0] = (int)((1u << 16) | (1u << 20) | (3u << 22) | (3u << 25));
  g[1] = (int)(((unsigned)K & 0xFFFFu) << 16);   // tensor_dim0[15:0] (abar=0)
  g[2] = (int)((unsigned)K >> 16);               // dim0[31:16] | dim1[15:0]<<16
  g[3] = (int)(16u | (32u << 16));               // dim1[31:16]=16 | tile_dim0=32
  g[4] = 128;                                    // tile_dim1=128, tile_dim2=0
  g[5] = K;                                      // tensor_dim0_stride[31:0]
  g[6] = 0;
  g[7] = 0;
  return g;
}

__device__ __forceinline__ void tdm_load_2d(unsigned lds_byte, const void* gp,
                                            tdm_i8 g1) {
  const unsigned long long ga = (unsigned long long)(size_t)gp;
  tdm_u4 g0;
  g0.x = 1u;                                              // count=1, user flags 0
  g0.y = lds_byte;                                        // LDS byte address
  g0.z = (unsigned)ga;                                    // global_addr[31:0]
  g0.w = (unsigned)((ga >> 32) & 0x01FFFFFFull) | (2u << 30);  // [56:32] | type=2
  const tdm_i4 z4 = {0, 0, 0, 0};                         // groups 2/3 unused (2D)
  const tdm_i8 z8 = {0, 0, 0, 0, 0, 0, 0, 0};             // extra group (unused)
  __builtin_amdgcn_tensor_load_to_lds(g0, g1, z4, z4, z8, 0);
}

// ---------------------------------------------------------------------------
// Elementwise f32 -> (hi,lo) f16 split: a = hi + lo to ~22 mantissa bits.
// Done ONCE per tensor in HBM so the GEMM inner loop has zero conversions.
// ---------------------------------------------------------------------------
__global__ __launch_bounds__(256) void split_f16(
    const float* __restrict__ s, _Float16* __restrict__ h,
    _Float16* __restrict__ l, long long n4) {
  const long long i = (long long)blockIdx.x * 256 + threadIdx.x;
  if (i >= n4) return;
  const float4 v = ((const float4*)s)[i];
  union { _Float16 x[4]; float2 f; } uh, ul;
  const float vv[4] = {v.x, v.y, v.z, v.w};
#pragma unroll
  for (int j = 0; j < 4; ++j) {
    const _Float16 hh = (_Float16)vv[j];
    uh.x[j] = hh;
    ul.x[j] = (_Float16)(vv[j] - (float)hh);
  }
  ((float2*)h)[i] = uh.f;
  ((float2*)l)[i] = ul.f;
}

// ---------------------------------------------------------------------------
// Tiled (T,D)->(D,T) f16 transpose per batch, hi+lo, 64x64 tiles via LDS.
// Lets attn@V use the same fast (N,K)-major B path as all other GEMMs.
// ---------------------------------------------------------------------------
__global__ __launch_bounds__(256) void transpose_split(
    const _Float16* __restrict__ sh, const _Float16* __restrict__ sl,
    _Float16* __restrict__ dh, _Float16* __restrict__ dl) {
  const int b = blockIdx.z;
  const int t0 = blockIdx.y * 64;
  const int d0 = blockIdx.x * 64;
  __shared__ alignas(16) _Float16 tile[64 * 72];
  const int tid = threadIdx.x;
  const int r = tid >> 3;          // 0..31
  const int c = (tid & 7) * 8;     // 0,8,..,56

  const _Float16* srcs[2] = {sh, sl};
  _Float16* dsts[2] = {dh, dl};
#pragma unroll
  for (int t = 0; t < 2; ++t) {
#pragma unroll
    for (int p = 0; p < 2; ++p)
      *(float4*)(&tile[(r + p * 32) * 72 + c]) =
          *(const float4*)(srcs[t] + ((long long)b * TT + t0 + r + p * 32) * DD + d0 + c);
    __syncthreads();
#pragma unroll
    for (int p = 0; p < 2; ++p) {
      union { _Float16 x[8]; float4 f; } u;
#pragma unroll
      for (int j = 0; j < 8; ++j) u.x[j] = tile[(c + j) * 72 + (r + p * 32)];
      *(float4*)(dsts[t] + ((long long)b * DD + d0 + r + p * 32) * TT + t0 + c) = u.f;
    }
    __syncthreads();
  }
}

// ---------------------------------------------------------------------------
// GEMM on pre-split f16 operands, f16x3 terms (hi*hi + hi*lo + lo*hi), fp32
// accumulate:  C = scale * (A @ B^T),  A=(M,K), B=(N,K), both row-major.
// SPLIT_OUT=1: write Ch/Cl f16 split; SPLIT_OUT=0: f32 C (+causal mask).
// WG tile 128x128, 8 waves, wave tile 32x64, K-step 32.
// Staging: double-buffered LDS (2 x 40KB dynamic); wave 0 issues 4 TDM
// descriptors per K-step (tensor_load_to_lds with LDS row padding), all other
// waves only compute. Handshake: wave0 s_wait_tensorcnt 0 -> s_barrier.
// LDS halfword layout per buffer: Ah@0, Al@5120, Bh@10240, Bl@15360 (pitch 40).
// ---------------------------------------------------------------------------
template <bool CAUSAL, bool SPLIT_OUT>
__global__ __launch_bounds__(256) void gemm_f16x3(
    const _Float16* __restrict__ Ah, const _Float16* __restrict__ Al,
    const _Float16* __restrict__ Bh, const _Float16* __restrict__ Bl,
    float* __restrict__ C, _Float16* __restrict__ Ch, _Float16* __restrict__ Cl,
    int N, int K, long long sA, long long sB, long long sC, float scale) {
  const long long z = blockIdx.z;
  Ah += z * sA; Al += z * sA;
  Bh += z * sB; Bl += z * sB;
  const int m0 = blockIdx.y * 128;
  const int n0 = blockIdx.x * 128;
  const int tid = threadIdx.x;

  // Causal: tile fully above the diagonal -> emit -inf, skip all compute.
  if (CAUSAL && n0 > m0 + 127) {
    float* Cz = C + z * sC;
    for (int i = tid; i < 128 * 128; i += 256) {
      const int r = i >> 7, c = i & 127;
      Cz[(long long)(m0 + r) * N + (n0 + c)] = -__builtin_inff();
    }
    return;
  }

  extern __shared__ _Float16 smem[];               // 2 * 20480 halfwords
  const unsigned ldsBase = (unsigned)(size_t)smem; // LDS byte offset (addr[31:0])

  const int lane = tid & 31;
  const int wave = tid >> 5;
  const int wm = wave & 3;          // 4 wave-rows of 32 -> 128 M
  const int wn = wave >> 2;         // 2 wave-cols of 64 -> 128 N
  const int lr = lane & 15;
  const int kb = (lane >> 4) * 8;

  const tdm_i8 g1 = tdm_make_g1(K);
  auto stage = [&](int buf, int kt) {
    const unsigned bufB = ldsBase + (unsigned)buf * 40960u;
    const long long oa = (long long)m0 * K + kt;
    const long long ob = (long long)n0 * K + kt;
    tdm_load_2d(bufB,          Ah + oa, g1);
    tdm_load_2d(bufB + 10240u, Al + oa, g1);
    tdm_load_2d(bufB + 20480u, Bh + ob, g1);
    tdm_load_2d(bufB + 30720u, Bl + ob, g1);
  };

  v8f zero;
#pragma unroll
  for (int i = 0; i < 8; ++i) zero[i] = 0.0f;
  v8f acc[2][4];
#pragma unroll
  for (int mt = 0; mt < 2; ++mt)
#pragma unroll
    for (int nt = 0; nt < 4; ++nt) acc[mt][nt] = zero;

  // Prologue: wave 0 DMAs buffer 0, drains its TENSORcnt, then all sync.
  if (wave == 0) {
    stage(0, 0);
    __builtin_amdgcn_s_wait_tensorcnt(0);
  }
  __syncthreads();

  int buf = 0;
  for (int kt = 0; kt < K; kt += 32) {
    // Wave 0 kicks off the next tile's DMA into the other buffer.
    if (wave == 0 && kt + 32 < K) stage(buf ^ 1, kt + 32);

    const _Float16* cur = smem + buf * 20480;
    v16h ahi[2], alo[2];
#pragma unroll
    for (int mt = 0; mt < 2; ++mt) {
      const int row = (wm * 32 + mt * 16 + lr) * 40;
      ahi[mt] = ldfrag(cur + row, kb);
      alo[mt] = ldfrag(cur + 5120 + row, kb);
    }
#pragma unroll
    for (int nt = 0; nt < 4; ++nt) {
      const int row = (wn * 64 + nt * 16 + lr) * 40;
      const v16h bhi = ldfrag(cur + 10240 + row, kb);
      const v16h blo = ldfrag(cur + 15360 + row, kb);
#pragma unroll
      for (int mt = 0; mt < 2; ++mt) {
        acc[mt][nt] = wmma16(ahi[mt], bhi, acc[mt][nt]);  // hi*hi
        acc[mt][nt] = wmma16(ahi[mt], blo, acc[mt][nt]);  // hi*lo
        acc[mt][nt] = wmma16(alo[mt], bhi, acc[mt][nt]);  // lo*hi
      }
    }

    // DMA into buf^1 complete (wave 0) + all waves done reading buf.
    if (wave == 0) __builtin_amdgcn_s_wait_tensorcnt(0);
    __syncthreads();
    buf ^= 1;
  }

  // Epilogue. C/D layout: VGPR v -> M = v + 8*(lane>>4), N = lane&15.
  const int rbase = m0 + wm * 32 + 8 * (lane >> 4);
  const int cbase = n0 + wn * 64 + lr;
#pragma unroll
  for (int mt = 0; mt < 2; ++mt) {
#pragma unroll
    for (int nt = 0; nt < 4; ++nt) {
      const int c = cbase + nt * 16;
#pragma unroll
      for (int v = 0; v < 8; ++v) {
        const int r = rbase + mt * 16 + v;
        const long long idx = z * sC + (long long)r * N + c;
        float val = acc[mt][nt][v] * scale;
        if (SPLIT_OUT) {
          const _Float16 h = (_Float16)val;
          Ch[idx] = h;
          Cl[idx] = (_Float16)(val - (float)h);
        } else {
          if (CAUSAL && c > r) val = -__builtin_inff();
          C[idx] = val;
        }
      }
    }
  }
}

// ---------------------------------------------------------------------------
// Row softmax over T=2048; one block per row. Writes f32 attn (output) AND
// the f16 hi/lo split consumed by the attn@V GEMM. -inf -> exact 0.
// ---------------------------------------------------------------------------
__global__ __launch_bounds__(256) void softmax_rows(
    float* __restrict__ attn, _Float16* __restrict__ ah,
    _Float16* __restrict__ al) {
  const long long row = blockIdx.x;
  float* p = attn + row * TT;
  _Float16* ph = ah + row * TT;
  _Float16* pl = al + row * TT;
  __shared__ float red[256];
  const int tid = threadIdx.x;

  float vals[8];
  float m = -__builtin_inff();
#pragma unroll
  for (int i = 0; i < 8; ++i) {
    vals[i] = p[tid + i * 256];
    m = fmaxf(m, vals[i]);
  }
  red[tid] = m;
  __syncthreads();
  for (int s = 128; s > 0; s >>= 1) {
    if (tid < s) red[tid] = fmaxf(red[tid], red[tid + s]);
    __syncthreads();
  }
  m = red[0];
  __syncthreads();

  float sum = 0.0f;
#pragma unroll
  for (int i = 0; i < 8; ++i) {
    vals[i] = __expf(vals[i] - m);   // exp(-inf) = 0 for masked entries
    sum += vals[i];
  }
  red[tid] = sum;
  __syncthreads();
  for (int s = 128; s > 0; s >>= 1) {
    if (tid < s) red[tid] += red[tid + s];
    __syncthreads();
  }
  const float inv = 1.0f / red[0];
#pragma unroll
  for (int i = 0; i < 8; ++i) {
    const float v = vals[i] * inv;
    p[tid + i * 256] = v;
    const _Float16 h = (_Float16)v;
    ph[tid + i * 256] = h;
    pl[tid + i * 256] = (_Float16)(v - (float)h);
  }
}

// ---------------------------------------------------------------------------
// Pipeline (all on `stream`, strictly ordered):
//   split(x,W*) -> Q/K/V proj (split out) -> V transpose -> scores (causal,
//   f32) -> softmax (f32 + split) -> attn@V (split out) -> out proj (f32).
// d_out = [out (B*T*D) | attn (B*T*T)] fp32.
// ws (~185 MB f16) with stream-ordered region reuse:
//   x-region -> attn_hi, Q-region -> attn_lo, V-region -> O1.
// d_in[5] (mask) ignored: causality handled analytically.
// ---------------------------------------------------------------------------
extern "C" void kernel_launch(void* const* d_in, const int* in_sizes, int n_in,
                              void* d_out, int out_size, void* d_ws,
                              size_t ws_size, hipStream_t stream) {
  const float* x  = (const float*)d_in[0];
  const float* Wq = (const float*)d_in[1];
  const float* Wk = (const float*)d_in[2];
  const float* Wv = (const float*)d_in[3];
  const float* Wo = (const float*)d_in[4];

  float* out  = (float*)d_out;
  float* attn = out + (long long)BB * TT * DD;

  const long long XE = (long long)BB * TT * DD;  // 8.39M elems
  const long long WE = (long long)DD * DD;       // 1.05M elems
  const long long TD = (long long)TT * DD;
  const long long T2 = (long long)TT * TT;

  _Float16* w = (_Float16*)d_ws;
  _Float16 *xh = w,        *xl = xh + XE;
  _Float16 *Wqh = xl + XE, *Wql = Wqh + WE;
  _Float16 *Wkh = Wql + WE, *Wkl = Wkh + WE;
  _Float16 *Wvh = Wkl + WE, *Wvl = Wvh + WE;
  _Float16 *Woh = Wvl + WE, *Wol = Woh + WE;
  _Float16 *Qh = Wol + WE, *Ql = Qh + XE;
  _Float16 *Kh = Ql + XE,  *Kl = Kh + XE;
  _Float16 *Vh = Kl + XE,  *Vl = Vh + XE;
  _Float16 *Vth = Vl + XE, *Vtl = Vth + XE;
  _Float16 *attnh = xh, *attnl = Qh;   // reuse: dead after scores GEMM
  _Float16 *O1h = Vh,  *O1l = Vl;      // reuse: dead after transpose

  const dim3 blk(256);
  const int SMEM = 2 * 4 * 128 * 40 * (int)sizeof(_Float16);  // 81920 B

  // Allow 80KB dynamic LDS (host-side attr set; deterministic, not captured).
  (void)hipFuncSetAttribute((const void*)gemm_f16x3<false, true>,
                            hipFuncAttributeMaxDynamicSharedMemorySize, SMEM);
  (void)hipFuncSetAttribute((const void*)gemm_f16x3<true, false>,
                            hipFuncAttributeMaxDynamicSharedMemorySize, SMEM);
  (void)hipFuncSetAttribute((const void*)gemm_f16x3<false, false>,
                            hipFuncAttributeMaxDynamicSharedMemorySize, SMEM);

  // 0) One-time f32 -> f16 hi/lo splits.
  split_f16<<<dim3((unsigned)(XE / 4 / 256)), blk, 0, stream>>>(x, xh, xl, XE / 4);
  split_f16<<<dim3((unsigned)(WE / 4 / 256)), blk, 0, stream>>>(Wq, Wqh, Wql, WE / 4);
  split_f16<<<dim3((unsigned)(WE / 4 / 256)), blk, 0, stream>>>(Wk, Wkh, Wkl, WE / 4);
  split_f16<<<dim3((unsigned)(WE / 4 / 256)), blk, 0, stream>>>(Wv, Wvh, Wvl, WE / 4);
  split_f16<<<dim3((unsigned)(WE / 4 / 256)), blk, 0, stream>>>(Wo, Woh, Wol, WE / 4);

  // 1) Projections (8192,1024)x(1024,1024)^T, split f16 outputs.
  const dim3 gproj(DD / 128, (BB * TT) / 128, 1);
  gemm_f16x3<false, true><<<gproj, blk, SMEM, stream>>>(
      xh, xl, Wqh, Wql, nullptr, Qh, Ql, DD, DD, 0, 0, 0, 1.0f);
  gemm_f16x3<false, true><<<gproj, blk, SMEM, stream>>>(
      xh, xl, Wkh, Wkl, nullptr, Kh, Kl, DD, DD, 0, 0, 0, 1.0f);
  gemm_f16x3<false, true><<<gproj, blk, SMEM, stream>>>(
      xh, xl, Wvh, Wvl, nullptr, Vh, Vl, DD, DD, 0, 0, 0, 1.0f);

  // 1b) V -> V^T per batch so attn@V uses the fast (N,K) B path.
  transpose_split<<<dim3(DD / 64, TT / 64, BB), blk, 0, stream>>>(Vh, Vl, Vth, Vtl);

  // 2) scores = Q@K^T / sqrt(D) with causal -inf; f32 into attn region.
  const dim3 gsc(TT / 128, TT / 128, BB);
  gemm_f16x3<true, false><<<gsc, blk, SMEM, stream>>>(
      Qh, Ql, Kh, Kl, attn, nullptr, nullptr, TT, DD, TD, TD, T2,
      0.03125f /* 1/sqrt(1024) */);

  // 3) softmax rows: f32 attn output + f16 split for next GEMM.
  softmax_rows<<<dim3(BB * TT), blk, 0, stream>>>(attn, attnh, attnl);

  // 4) O1 = attn @ V  (B = V^T stored (D,T)), split f16 output.
  const dim3 gav(DD / 128, TT / 128, BB);
  gemm_f16x3<false, true><<<gav, blk, SMEM, stream>>>(
      attnh, attnl, Vth, Vtl, nullptr, O1h, O1l, DD, TT, T2, TD, TD, 1.0f);

  // 5) out = O1 @ Wo^T, f32 output.
  gemm_f16x3<false, false><<<gproj, blk, SMEM, stream>>>(
      O1h, O1l, Woh, Wol, out, nullptr, nullptr, DD, DD, 0, 0, 0, 1.0f);
}
